// SequencialDecoder_59992103190822
// MI455X (gfx1250) — matrix operand order repaired and checked by reading, not compiled, not run on hardware
//
#include <hip/hip_runtime.h>

#define BB 1024
#define NN 100
#define HH 256
#define CLIPV 10.0f

typedef __attribute__((ext_vector_type(16))) __bf16        v16bf;
typedef __attribute__((ext_vector_type(8)))  float         v8f;
typedef __attribute__((ext_vector_type(4)))  unsigned int  v4u;

union Frag { v16bf v; v4u u[2]; };

// fast transcendentals: v_exp_f32 + v_rcp_f32 (TRANS pipe, no fdiv expansion)
__device__ __forceinline__ float ftanh(float x) {
    float cx = fminf(fmaxf(x, -15.f), 15.f);
    float e  = __expf(2.f * cx);
    return 1.f - 2.f * __builtin_amdgcn_rcpf(e + 1.f);
}
__device__ __forceinline__ float fsig(float x) {
    float cx = fminf(fmaxf(x, -30.f), 30.f);
    return __builtin_amdgcn_rcpf(1.f + __expf(-cx));
}

// ---------------------------------------------------------------------------
// Workspace layout (bytes), all 16B aligned:
//   W1p  [9][48][32][16] bf16 : 442368
//   W2p  [8][48][32][16] bf16 : 393216
//   Wqp  [8][16][32][16] bf16 : 131072
//   Wrp  [8][16][32][16] bf16 : 131072
//   refb [B*N*H]         bf16 : 52428800
#define OFF_W1  0
#define OFF_W2  442368
#define OFF_WQ  835584
#define OFF_WR  966656
#define OFF_REF 1097728

// ---------------------------------------------------------------------------
// Repack weights into WMMA B-fragment order:
//   flat = ((kt*NT + nt)*32 + lane)*16 + e
//   holds  B[kt*32 + (lane>=16?8:0) + (e&7) + (e>=8?16:0)][nt*16 + (lane&15)]
__global__ __launch_bounds__(256) void prep_weights(
    const float* __restrict__ W_ih1, const float* __restrict__ W_ih2,
    const float* __restrict__ W_q,   const float* __restrict__ W_ref,
    __bf16* __restrict__ W1p, __bf16* __restrict__ W2p,
    __bf16* __restrict__ Wqp, __bf16* __restrict__ Wrp)
{
    int idx = blockIdx.x * 256 + threadIdx.x;
    const int n1 = 9*48*512, n2 = 8*48*512, n3 = 8*16*512, n4 = 8*16*512;
    __bf16* dst; int NT, mode; int j = idx;
    if (j < n1)              { dst = W1p; NT = 48; mode = 0; }
    else if ((j -= n1) < n2) { dst = W2p; NT = 48; mode = 1; }
    else if ((j -= n2) < n3) { dst = Wqp; NT = 16; mode = 2; }
    else if ((j -= n3) < n4) { dst = Wrp; NT = 16; mode = 3; }
    else return;
    int e = j & 15, lane = (j >> 4) & 31, tile = j >> 9;
    int nt = tile % NT, kt = tile / NT;
    int k   = kt*32 + ((lane & 16) ? 8 : 0) + (e & 7) + ((e & 8) ? 16 : 0);
    int col = nt*16 + (lane & 15);
    float val;
    if (mode == 0)      val = (k < 257) ? W_ih1[col*257 + k] : 0.f;  // B[k][j]=W_ih1[j][k], pad K->288
    else if (mode == 1) val = W_ih2[col*256 + k];                    // B[k][j]=W_ih2[j][k]
    else if (mode == 2) val = W_q  [k*256 + col];                    // q = h2 @ W_q
    else                val = W_ref[k*256 + col];                    // ref = enc @ W_ref
    dst[j] = (__bf16)val;
}

// ---------------------------------------------------------------------------
// 16xN tile-row matmul: A (16 x 32*KT bf16 in LDS), B fragment-packed in
// global (L2-resident), D scattered to LDS f32. NWAVES waves, NITER tiles/wave.
template<int NT, int KT, int NITER, int NWAVES>
__device__ __forceinline__ void mm_phase(const __bf16* sA, int rowStrideBytes,
                                         const __bf16* __restrict__ Wp,
                                         float* sOut, int outStride,
                                         int lane, int wv)
{
    const int hi = (lane >> 4) & 1, ml = lane & 15;
    const char* arow = (const char*)sA + ml * rowStrideBytes;
    const int aoff = hi ? 16 : 0;
    #pragma unroll
    for (int it = 0; it < NITER; ++it) {
        const int nt = wv + NWAVES * it;
        v8f acc = {};
        #pragma unroll
        for (int kt = 0; kt < KT; ++kt) {
            Frag a, b;
            a.u[0] = *(const v4u*)(arow + kt*64 + aoff);        // K = kt*32 + off .. +7
            a.u[1] = *(const v4u*)(arow + kt*64 + 32 + aoff);   // K = kt*32 + 16 + off .. +7
            const v4u* bp = (const v4u*)((const char*)Wp +
                             (((size_t)(kt*NT + nt)*32 + lane) << 5));
            b.u[0] = bp[0]; b.u[1] = bp[1];
            acc = __builtin_amdgcn_wmma_f32_16x16x32_bf16(
                      false, a.v, false, b.v, (short)0, acc, false, false);
        }
        #pragma unroll
        for (int r = 0; r < 8; ++r)
            sOut[(r + hi*8)*outStride + nt*16 + ml] = acc[r];   // D: m = r + hi*8, n = lane&15
    }
}

// ---------------------------------------------------------------------------
// ref[b,n,:] = enc[b,n,:] @ W_ref  -> bf16. One block per 16 flattened rows.
__global__ __launch_bounds__(256) void ref_kernel(
    const float* __restrict__ enc, const __bf16* __restrict__ Wrp,
    __bf16* __restrict__ refb)
{
    __shared__ __align__(16) __bf16 s_a[16][256];
    const int tid = threadIdx.x, lane = tid & 31, wv = tid >> 5;
    const int hi = (lane >> 4) & 1, ml = lane & 15;
    const size_t row0 = (size_t)blockIdx.x * 16;
    for (int i = 0; i < 16; ++i)
        s_a[i][tid] = (__bf16)enc[(row0 + i) * HH + tid];
    __syncthreads();
    const char* arow = (const char*)&s_a[ml][0];
    const int aoff = hi ? 16 : 0;
    #pragma unroll
    for (int jj = 0; jj < 2; ++jj) {
        const int nt = wv + 8*jj;
        v8f acc = {};
        #pragma unroll
        for (int kt = 0; kt < 8; ++kt) {
            Frag a, b;
            a.u[0] = *(const v4u*)(arow + kt*64 + aoff);
            a.u[1] = *(const v4u*)(arow + kt*64 + 32 + aoff);
            const v4u* bp = (const v4u*)((const char*)Wrp +
                             (((size_t)(kt*16 + nt)*32 + lane) << 5));
            b.u[0] = bp[0]; b.u[1] = bp[1];
            acc = __builtin_amdgcn_wmma_f32_16x16x32_bf16(
                      false, a.v, false, b.v, (short)0, acc, false, false);
        }
        #pragma unroll
        for (int r = 0; r < 8; ++r)
            refb[(row0 + r + hi*8) * HH + nt*16 + ml] = (__bf16)acc[r];
    }
}

// ---------------------------------------------------------------------------
// Persistent decoder: one block = 16 batch rows, 16 waves, full 100-step loop.
__global__ __launch_bounds__(512) void decode_kernel(
    const float* __restrict__ x,     const float* __restrict__ enc,
    const float* __restrict__ b_ih1, const float* __restrict__ b_hh1,
    const float* __restrict__ b_ih2, const float* __restrict__ b_hh2,
    const float* __restrict__ vvec,
    const __bf16* __restrict__ W1p, const __bf16* __restrict__ W2p,
    const __bf16* __restrict__ Wqp, const __bf16* __restrict__ refb,
    float* __restrict__ out)
{
    __shared__ __align__(16) __bf16 s_ctx[16][288];   // [enc row | D | zero pad]
    __shared__ __align__(16) __bf16 s_h[16][256];     // h1 then h2
    __shared__ __align__(16) float  s_gi[16][768];
    __shared__ __align__(16) float  s_q[16][256];
    __shared__ float    s_u[16][104];
    __shared__ unsigned s_mask[16][4];
    __shared__ int      s_nxt[16];
    __shared__ float    s_Dv[16];
    __shared__ float    s_v[256];

    const int tid  = threadIdx.x;
    const int lane = tid & 31, wv = tid >> 5;      // 16 waves
    const int jc   = tid & 255;                    // column for gate/ctx phases
    const int bb0  = blockIdx.x * 16;

    if (tid < 256) s_v[tid] = vvec[tid];
    // ctx0 = [mean_n enc[b], 1.0]
    for (int p = tid; p < 4096; p += 512) {
        const int m = p >> 8;
        const float* ep = enc + ((size_t)(bb0 + m) * NN) * HH + jc;
        float s = 0.f;
        for (int n = 0; n < NN; ++n) s += ep[(size_t)n * HH];
        s_ctx[m][jc] = (__bf16)(s * (1.0f / NN));
    }
    if (tid < 16) {
        s_ctx[tid][256] = (__bf16)1.0f;
        for (int k2 = 257; k2 < 288; ++k2) s_ctx[tid][k2] = (__bf16)0.f;
        s_mask[tid][0] = s_mask[tid][1] = s_mask[tid][2] = s_mask[tid][3] = 0u;
    }
    __syncthreads();

    // gh = b_hh (hidden state is 0 in reference) -> W_hh matmuls vanish
    const float b1r  = b_ih1[jc]     + b_hh1[jc];
    const float b1z  = b_ih1[256+jc] + b_hh1[256+jc];
    const float b1n  = b_ih1[512+jc];
    const float b1hn = b_hh1[512+jc];
    const float b2r  = b_ih2[jc]     + b_hh2[jc];
    const float b2z  = b_ih2[256+jc] + b_hh2[256+jc];
    const float b2n  = b_ih2[512+jc];
    const float b2hn = b_hh2[512+jc];

    float ll = 0.f, cost = 0.f, Dv = 1.0f;          // live in lane 0 of wave wv
    float fx = 0.f, fy = 0.f, px = 0.f, py = 0.f;

    for (int t = 0; t < NN; ++t) {
        // GI1 = ctx @ W_ih1^T   (16 x 768, K = 288 padded)
        mm_phase<48, 9, 3, 16>(&s_ctx[0][0], 288*2, W1p, &s_gi[0][0], 768, lane, wv);
        __syncthreads();
        for (int p = tid; p < 4096; p += 512) {      // gates -> h1 = (1-z)*n
            const int m = p >> 8;
            float r   = fsig(s_gi[m][jc] + b1r);
            float z   = fsig(s_gi[m][256+jc] + b1z);
            float nn2 = ftanh(s_gi[m][512+jc] + b1n + r * b1hn);
            s_h[m][jc] = (__bf16)((1.f - z) * nn2);
        }
        __syncthreads();
        // GI2 = h1 @ W_ih2^T
        mm_phase<48, 8, 3, 16>(&s_h[0][0], 256*2, W2p, &s_gi[0][0], 768, lane, wv);
        __syncthreads();
        for (int p = tid; p < 4096; p += 512) {      // gates -> h2
            const int m = p >> 8;
            float r   = fsig(s_gi[m][jc] + b2r);
            float z   = fsig(s_gi[m][256+jc] + b2z);
            float nn2 = ftanh(s_gi[m][512+jc] + b2n + r * b2hn);
            s_h[m][jc] = (__bf16)((1.f - z) * nn2);
        }
        __syncthreads();
        // q = h2 @ W_q
        mm_phase<16, 8, 1, 16>(&s_h[0][0], 256*2, Wqp, &s_q[0][0], 256, lane, wv);
        __syncthreads();
        // attention: u[m][n] = CLIP*tanh( sum_h tanh(q+ref)*v ), masked
        for (int p = tid; p < 16 * NN; p += 512) {
            int m = p / NN, n = p - m * NN;
            float uu;
            if ((s_mask[m][n >> 5] >> (n & 31)) & 1u) uu = -1.0e9f;
            else {
                const unsigned* rp =
                    (const unsigned*)(refb + ((size_t)(bb0 + m) * NN + n) * HH);
                const float* qm = &s_q[m][0];
                float s = 0.f;
                for (int hw = 0; hw < 128; ++hw) {
                    unsigned pk = rp[hw];                       // 2 x bf16
                    float r0 = __uint_as_float(pk << 16);
                    float r1 = __uint_as_float(pk & 0xffff0000u);
                    s += ftanh(qm[2*hw]   + r0) * s_v[2*hw]
                       + ftanh(qm[2*hw+1] + r1) * s_v[2*hw+1];
                }
                uu = CLIPV * ftanh(s);
            }
            s_u[m][n] = uu;
        }
        __syncthreads();
        // wave-parallel argmax + log_softmax: wave wv owns row wv
        {
            float bv = -2.0e9f; int bi = NN;
            for (int n = lane; n < NN; n += 32) {
                float uv = s_u[wv][n];
                if (uv > bv) { bv = uv; bi = n; }   // first-max within lane subset
            }
            #pragma unroll
            for (int sh = 16; sh >= 1; sh >>= 1) {  // value-max, tie -> smaller index
                float ov = __shfl_xor(bv, sh, 32);
                int   oi = __shfl_xor(bi, sh, 32);
                if (ov > bv || (ov == bv && oi < bi)) { bv = ov; bi = oi; }
            }
            float se = 0.f;
            for (int n = lane; n < NN; n += 32) se += __expf(s_u[wv][n] - bv);
            #pragma unroll
            for (int sh = 16; sh >= 1; sh >>= 1) se += __shfl_xor(se, sh, 32);
            if (lane == 0) {
                ll -= __logf(se);                   // u[bi]==bv (argmax)
                s_mask[wv][bi >> 5] |= (1u << (bi & 31));
                Dv -= 1.0f / NN;
                s_Dv[wv] = Dv;
                s_nxt[wv] = bi;
                const float* cxy = x + ((size_t)(bb0 + wv) * NN + bi) * 2;
                float cx = cxy[0], cy = cxy[1];
                if (t == 0) { fx = cx; fy = cy; }
                else { float dx = px - cx, dy = py - cy; cost += sqrtf(dx*dx + dy*dy); }
                px = cx; py = cy;
            }
        }
        __syncthreads();
        // ctx <- [enc[b, nxt], D]
        for (int p = tid; p < 4096; p += 512) {
            const int m = p >> 8;
            s_ctx[m][jc] = (__bf16)enc[((size_t)(bb0 + m) * NN + s_nxt[m]) * HH + jc];
        }
        if (tid < 16) s_ctx[tid][256] = (__bf16)s_Dv[tid];
        __syncthreads();
    }
    if (lane == 0) {                                // close the tour; wave wv = row wv
        float dx = px - fx, dy = py - fy;
        cost += sqrtf(dx*dx + dy*dy);
        out[bb0 + wv]      = cost;
        out[BB + bb0 + wv] = ll;
    }
}

// ---------------------------------------------------------------------------
extern "C" void kernel_launch(void* const* d_in, const int* in_sizes, int n_in,
                              void* d_out, int out_size, void* d_ws, size_t ws_size,
                              hipStream_t stream)
{
    (void)in_sizes; (void)n_in; (void)out_size; (void)ws_size;
    const float* x     = (const float*)d_in[0];
    const float* enc   = (const float*)d_in[1];
    const float* W_ih1 = (const float*)d_in[2];
    /* d_in[3] = W_hh1: unused (hidden state is 0 -> only b_hh matters) */
    const float* b_ih1 = (const float*)d_in[4];
    const float* b_hh1 = (const float*)d_in[5];
    const float* W_ih2 = (const float*)d_in[6];
    /* d_in[7] = W_hh2: unused */
    const float* b_ih2 = (const float*)d_in[8];
    const float* b_hh2 = (const float*)d_in[9];
    const float* W_q   = (const float*)d_in[10];
    const float* W_ref = (const float*)d_in[11];
    const float* v     = (const float*)d_in[12];

    char* ws = (char*)d_ws;
    __bf16* W1p  = (__bf16*)(ws + OFF_W1);
    __bf16* W2p  = (__bf16*)(ws + OFF_W2);
    __bf16* Wqp  = (__bf16*)(ws + OFF_WQ);
    __bf16* Wrp  = (__bf16*)(ws + OFF_WR);
    __bf16* refb = (__bf16*)(ws + OFF_REF);

    prep_weights<<<2144, 256, 0, stream>>>(W_ih1, W_ih2, W_q, W_ref,
                                           W1p, W2p, Wqp, Wrp);
    ref_kernel<<<(BB * NN) / 16, 256, 0, stream>>>(enc, Wrp, refb);
    decode_kernel<<<BB / 16, 512, 0, stream>>>(x, enc,
                                               b_ih1, b_hh1, b_ih2, b_hh2, v,
                                               W1p, W2p, Wqp, refb,
                                               (float*)d_out);
}
